// NTXentLoss_7902739824787
// MI455X (gfx1250) — matrix operand order repaired and checked by reading, not compiled, not run on hardware
//
#include <hip/hip_runtime.h>
#include <hip/hip_bf16.h>

typedef __attribute__((ext_vector_type(16))) _Float16 v16h;
typedef __attribute__((ext_vector_type(8)))  _Float16 v8h;
typedef __attribute__((ext_vector_type(4)))  _Float16 v4h;
typedef __attribute__((ext_vector_type(8)))  float    v8f;

#define DIMS   512
#define NROWS  8192
#define BHALF  4096
#define TAU    0.07f
#define EPSN   1e-8f

// ---------------------------------------------------------------------------
// Kernel 1: row L2-normalize concat(anchor, positive) -> f16 z in workspace.
// One 128-thread block per row; each thread handles 4 contiguous floats.
// ---------------------------------------------------------------------------
__global__ void ntx_normalize(const float* __restrict__ anchor,
                              const float* __restrict__ positive,
                              _Float16* __restrict__ z) {
    const int row = blockIdx.x;                 // 0..8191
    const int t   = threadIdx.x;                // 0..127
    const float* src = (row < BHALF) ? (anchor + (size_t)row * DIMS)
                                     : (positive + (size_t)(row - BHALF) * DIMS);
    const float4 x = ((const float4*)src)[t];
    float ss = x.x * x.x + x.y * x.y + x.z * x.z + x.w * x.w;
    // wave32 reduce
    #pragma unroll
    for (int off = 16; off; off >>= 1) ss += __shfl_xor(ss, off, 32);
    __shared__ float parts[4];
    if ((t & 31) == 0) parts[t >> 5] = ss;
    __syncthreads();
    const float total = parts[0] + parts[1] + parts[2] + parts[3];
    const float inv = 1.0f / fmaxf(sqrtf(total), EPSN);
    v4h h;
    h[0] = (_Float16)(x.x * inv);
    h[1] = (_Float16)(x.y * inv);
    h[2] = (_Float16)(x.z * inv);
    h[3] = (_Float16)(x.w * inv);
    *(v4h*)(z + (size_t)row * DIMS + 4 * t) = h;
}

// ---------------------------------------------------------------------------
// Kernel 2: fused sim = (z z^T)/tau with row-wise logsumexp (fixed shift
// M = 1/tau, valid since cosine <= 1). One wave per 16-row stripe; A kept
// resident in 16 v16h fragments (128 VGPRs); B streamed from L2 (z is 8MB).
// waves_per_eu(1) gives the RA the full register file -> no scratch spills.
// ---------------------------------------------------------------------------
__global__ __launch_bounds__(128)
__attribute__((amdgpu_waves_per_eu(1)))
void ntx_sim_lse(const _Float16* __restrict__ z,
                 float* __restrict__ loss) {
    const int lane   = threadIdx.x & 31;
    const int wave   = threadIdx.x >> 5;
    const int lo     = lane & 15;       // column-within-tile / row-within-tile
    const int hi     = lane >> 4;       // half-wave select
    const int stripe = blockIdx.x * (blockDim.x >> 5) + wave;
    const int r0     = stripe * 16;
    const int igbase = r0 + 8 * hi;     // global row of this lane's C vgpr v=0

    const float INV_TAU = 1.0f / TAU;
    const float M = INV_TAU;            // logsumexp shift bound (cosine <= 1)

    // --- Load A fragments (contiguous K order, same as B) once: 16 K-tiles ---
    // lanes 0-15:  elems 0..15 -> K = kb+0..15
    // lanes 16-31: elems 0..15 -> K = kb+16..31
    v16h afrag[16];
    const _Float16* arow = z + (size_t)(r0 + lo) * DIMS;
    #pragma unroll
    for (int kt = 0; kt < 16; ++kt) {
        const int kb = kt * 32;
        const v8h a_lo = *(const v8h*)(arow + kb + 16 * hi);
        const v8h a_hi = *(const v8h*)(arow + kb + 16 * hi + 8);
        v16h a;
        #pragma unroll
        for (int e = 0; e < 8; ++e) { a[e] = a_lo[e]; a[e + 8] = a_hi[e]; }
        afrag[kt] = a;
    }

    // Per-lane logsumexp state for 8 rows
    // (C/D layout: lane L, vgpr v -> row M = v + 8*(L/16), col N = L%16)
    float sm[8], pos[8];
    #pragma unroll
    for (int v = 0; v < 8; ++v) { sm[v] = 0.0f; pos[v] = 0.0f; }

    for (int jb = 0; jb < NROWS; jb += 16) {
        // B fragment (contiguous K order, same as A): lane holds column n = lane%16,
        // K = kb + 16*(lane/16) .. +15, contiguous in row-major z.
        const _Float16* brow = z + (size_t)(jb + lo) * DIMS + 16 * hi;
        __builtin_prefetch(z + (size_t)(((jb + 16) & (NROWS - 1)) + lo) * DIMS, 0, 3);

        v8f c0 = {}, c1 = {};
        #pragma unroll
        for (int kt = 0; kt < 16; kt += 2) {
            const v16h b0 = *(const v16h*)(brow + kt * 32);
            const v16h b1 = *(const v16h*)(brow + kt * 32 + 32);
            c0 = __builtin_amdgcn_wmma_f32_16x16x32_f16(false, afrag[kt],     false, b0,
                                                        (short)0, c0, false, false);
            c1 = __builtin_amdgcn_wmma_f32_16x16x32_f16(false, afrag[kt + 1], false, b1,
                                                        (short)0, c1, false, false);
        }

        // Diagonal / positive detection without per-row index arrays:
        // jg == ig(v)       <=>  jg - igbase == v
        // jg == (ig+B)%N    <=>  ((jg+B)&(N-1)) - igbase == v   (involution)
        const int jg     = jb + lo;
        const int d_self = jg - igbase;
        const int d_pos  = ((jg + BHALF) & (NROWS - 1)) - igbase;
        #pragma unroll
        for (int v = 0; v < 8; ++v) {
            const float val = (c0[v] + c1[v]) * INV_TAU;      // sim(ig+v, jg)
            const float e   = __expf(val - M);
            sm[v]  += (v == d_self) ? 0.0f : e;               // skip diagonal
            pos[v] += (v == d_pos)  ? val  : 0.0f;            // capture positive
        }
    }

    // Reduce the 16 column-lanes of each half-wave (xor masks stay in-half).
    #pragma unroll
    for (int v = 0; v < 8; ++v) {
        float s = sm[v], p = pos[v];
        #pragma unroll
        for (int off = 1; off < 16; off <<= 1) {
            s += __shfl_xor(s, off, 32);
            p += __shfl_xor(p, off, 32);
        }
        if (lo == 0) loss[igbase + v] = -p + M + __logf(s);
    }
}

// ---------------------------------------------------------------------------
// Kernel 3: deterministic mean of 8192 per-row losses.
// ---------------------------------------------------------------------------
__global__ void ntx_mean(const float* __restrict__ loss, float* __restrict__ out) {
    __shared__ float s[256];
    float acc = 0.0f;
    for (int i = threadIdx.x; i < NROWS; i += 256) acc += loss[i];
    s[threadIdx.x] = acc;
    __syncthreads();
    for (int w = 128; w; w >>= 1) {
        if (threadIdx.x < w) s[threadIdx.x] += s[threadIdx.x + w];
        __syncthreads();
    }
    if (threadIdx.x == 0) out[0] = s[0] / (float)NROWS;
}

extern "C" void kernel_launch(void* const* d_in, const int* in_sizes, int n_in,
                              void* d_out, int out_size, void* d_ws, size_t ws_size,
                              hipStream_t stream) {
    const float* anchor   = (const float*)d_in[0];
    const float* positive = (const float*)d_in[1];
    _Float16* z    = (_Float16*)d_ws;                                   // 8 MB
    float*    loss = (float*)((char*)d_ws + (size_t)NROWS * DIMS * 2);  // 32 KB

    ntx_normalize<<<NROWS, 128, 0, stream>>>(anchor, positive, z);
    // 512 stripes of 16 rows, 4 waves (128 threads) per block -> 128 blocks.
    ntx_sim_lse<<<128, 128, 0, stream>>>(z, loss);
    ntx_mean<<<1, 256, 0, stream>>>(loss, (float*)d_out);
}
